// FlexBertGLUMoE_28827820490912
// MI455X (gfx1250) — matrix operand order repaired and checked by reading, not compiled, not run on hardware
//
#include <hip/hip_runtime.h>
#include <hip/hip_bf16.h>
#include <math.h>

// ---------------- problem constants ----------------
constexpr int Bb = 4, Tt = 2048, Hh = 1024;
constexpr int Ee = 8, Kk = 2, Ff = 2048;
constexpr int Ss = Bb * Tt;          // 8192 tokens
constexpr int CAP = 2560;            // ceil(K*S/E*1.25), divisible by 256
constexpr int TWO_F = 2 * Ff;        // 4096

// ---------------- types for WMMA ----------------
typedef __attribute__((ext_vector_type(16))) __bf16 v16bf;
typedef __attribute__((ext_vector_type(8)))  float  v8f;

union Frag { v16bf v; uint4 q[2]; };

__device__ __forceinline__ unsigned short f32_to_bf16(float f) {
  unsigned int u = __float_as_uint(f);
  if ((u & 0x7fffffffu) > 0x7f800000u) return (unsigned short)((u >> 16) | 0x0040u); // quiet NaN
  u += 0x7fffu + ((u >> 16) & 1u);   // round-to-nearest-even
  return (unsigned short)(u >> 16);
}

// CDNA5 async global->LDS copy (ASYNCcnt path, bypasses VGPR staging).
// lds_addr = low 32 bits of the generic pointer.
__device__ __forceinline__ void async_copy16(unsigned lds_addr, const void* gsrc) {
  asm volatile("global_load_async_to_lds_b128 %0, %1, off"
               :: "v"(lds_addr), "v"((unsigned long long)(uintptr_t)gsrc)
               : "memory");
}
__device__ __forceinline__ void wait_async0() {
  asm volatile("s_wait_asynccnt 0x0" ::: "memory");
}

// ---------------- 0) zero workspace region (uint4 stores) ----------------
__global__ void zero_kernel(uint4* __restrict__ p, size_t n16) {
  size_t i = (size_t)blockIdx.x * blockDim.x + threadIdx.x;
  size_t stride = (size_t)gridDim.x * blockDim.x;
  uint4 z; z.x = z.y = z.z = z.w = 0u;
  for (; i < n16; i += stride) p[i] = z;
}

// ---------------- 1) router: logits -> softmax -> top2 ----------------
__global__ __launch_bounds__(256) void router_kernel(
    const float* __restrict__ x, const float* __restrict__ Wg,
    int* __restrict__ eArr, float* __restrict__ vArr) {
  const int wave = threadIdx.x >> 5;
  const int lane = threadIdx.x & 31;
  const int s = blockIdx.x * 8 + wave;
  const float* xr = x + (size_t)s * Hh;
  float acc[Ee];
#pragma unroll
  for (int e = 0; e < Ee; ++e) acc[e] = 0.f;
  for (int j = lane; j < Hh; j += 32) {
    float xv = xr[j];
    const float* wr = Wg + (size_t)j * Ee;
#pragma unroll
    for (int e = 0; e < Ee; ++e) acc[e] += xv * wr[e];
  }
#pragma unroll
  for (int off = 16; off >= 1; off >>= 1) {
#pragma unroll
    for (int e = 0; e < Ee; ++e) acc[e] += __shfl_xor(acc[e], off, 32);
  }
  if (lane == 0) {
    float m = acc[0];
#pragma unroll
    for (int e = 1; e < Ee; ++e) m = fmaxf(m, acc[e]);
    float p[Ee], sum = 0.f;
#pragma unroll
    for (int e = 0; e < Ee; ++e) { p[e] = expf(acc[e] - m); sum += p[e]; }
    float inv = 1.f / sum;
#pragma unroll
    for (int e = 0; e < Ee; ++e) p[e] *= inv;
    int i0 = 0;
#pragma unroll
    for (int e = 1; e < Ee; ++e) if (p[e] > p[i0]) i0 = e;
    int i1 = (i0 == 0) ? 1 : 0;
#pragma unroll
    for (int e = 0; e < Ee; ++e) if (e != i0 && p[e] > p[i1]) i1 = e;
    eArr[s]      = i0;  vArr[s]      = p[i0];
    eArr[Ss + s] = i1;  vArr[Ss + s] = p[i1];
  }
}

// ---------------- 2) deterministic capacity scan (matches jnp.cumsum order) -
__global__ __launch_bounds__(1024) void scan_kernel(
    const int* __restrict__ eArr, int* __restrict__ slotArr) {
  __shared__ int cnt[1024][Ee];
  const int tid = threadIdx.x;
  int local[Ee];
#pragma unroll
  for (int e = 0; e < Ee; ++e) local[e] = 0;
  const int base = tid * 16;
#pragma unroll
  for (int i = 0; i < 16; ++i) local[eArr[base + i]]++;
#pragma unroll
  for (int e = 0; e < Ee; ++e) cnt[tid][e] = local[e];
  __syncthreads();
  for (int off = 1; off < 1024; off <<= 1) {
    int tmp[Ee];
    if (tid >= off) {
#pragma unroll
      for (int e = 0; e < Ee; ++e) tmp[e] = cnt[tid - off][e];
    }
    __syncthreads();
    if (tid >= off) {
#pragma unroll
      for (int e = 0; e < Ee; ++e) cnt[tid][e] += tmp[e];
    }
    __syncthreads();
  }
  int run[Ee];
#pragma unroll
  for (int e = 0; e < Ee; ++e) run[e] = cnt[tid][e] - local[e]; // exclusive
#pragma unroll
  for (int i = 0; i < 16; ++i) {
    int e = eArr[base + i];
    slotArr[base + i] = run[e]++;
  }
}

// ---------------- 3) gather tokens into expert buffers (f32 -> bf16) -------
__global__ __launch_bounds__(256) void gather_kernel(
    const float* __restrict__ x, const int* __restrict__ eArr,
    const int* __restrict__ slotArr, unsigned short* __restrict__ inpB) {
  const int kk = blockIdx.x;
  const int slot = slotArr[kk];
  if (slot >= CAP) return;
  const int e = eArr[kk];
  const int s = (kk < Ss) ? kk : (kk - Ss);
  const float* src = x + (size_t)s * Hh;
  unsigned short* dst = inpB + ((size_t)e * CAP + slot) * Hh;
  for (int i = threadIdx.x; i < Hh; i += 256) dst[i] = f32_to_bf16(src[i]);
}

// ---------------- 4) transpose+cast weights: out[c][r] = bf16(in[r][c]) ----
__global__ __launch_bounds__(256) void transpose_cast_kernel(
    const float* __restrict__ in, unsigned short* __restrict__ out,
    int R, int Cc) {
  __shared__ float tile[32][33];
  const float* src = in + (size_t)blockIdx.z * R * Cc;
  unsigned short* dst = out + (size_t)blockIdx.z * R * Cc;
  const int c0 = blockIdx.x * 32, r0 = blockIdx.y * 32;
  const int tx = threadIdx.x, ty = threadIdx.y;   // (32, 8)
#pragma unroll
  for (int i = 0; i < 32; i += 8)
    tile[ty + i][tx] = src[(size_t)(r0 + ty + i) * Cc + c0 + tx];
  __syncthreads();
#pragma unroll
  for (int i = 0; i < 32; i += 8)
    dst[(size_t)(c0 + ty + i) * R + r0 + tx] = f32_to_bf16(tile[tx][ty + i]);
}

// ---------------- 5) WMMA BF16 GEMM (optionally fused GLU) ----------------
// A: [E][M][K] bf16 row-major; Bt: [E][rowsB][K] bf16 (N-major, K contiguous)
// Block tile 256 x (64|128); 8 waves, each 64x64 -> 16 WMMA : 16 ds_b128/K-step.
// Next K-tile prefetched via global_load_async_to_lds_b128 (ASYNCcnt).
#define BM 256
#define BK 32
#define LDK 56   // padded ushort row stride: 112 B = 28 banks -> conflict-free

template<bool GLU>
__global__ __launch_bounds__(256) void wmma_gemm_kernel(
    const unsigned short* __restrict__ Aall,
    const unsigned short* __restrict__ Ball,
    void* __restrict__ OutAll,
    int Kdim, int Nout, size_t strideA, size_t strideB, size_t strideO) {
  constexpr int BNcols = GLU ? 64 : 128;
  __shared__ unsigned short As[2][BM][LDK];   // 57344 B
  __shared__ unsigned short Bs[2][128][LDK];  // 28672 B
  constexpr unsigned AsBufBytes = BM * LDK * 2;   // 28672
  constexpr unsigned BsBufBytes = 128 * LDK * 2;  // 14336

  const int e = blockIdx.z;
  const unsigned short* A = Aall + (size_t)e * strideA;
  const unsigned short* B = Ball + (size_t)e * strideB;
  const int row0 = blockIdx.x * BM;
  const int col0 = blockIdx.y * BNcols;
  const int tid  = threadIdx.x;
  const int lane = tid & 31;
  const int wave = tid >> 5;
  const int wm0  = (wave & 3) * 64;                    // 4 M-waves x 64 rows
  const int wn0  = (wave >> 2) * (GLU ? 32 : 64);      // 2 N-waves
  const int half = lane >> 4;
  const int l15  = lane & 15;

  v8f acc[4][4];
  const v8f z8 = {0.f,0.f,0.f,0.f,0.f,0.f,0.f,0.f};
#pragma unroll
  for (int mi = 0; mi < 4; ++mi)
#pragma unroll
    for (int ni = 0; ni < 4; ++ni) acc[mi][ni] = z8;

  const int ksteps = Kdim / BK;

  // per-thread async-copy slots: 4 A chunks + 2 B chunks of 16 B per K-step
  const unsigned short* aSrc[4];
  unsigned aDst0[4];
#pragma unroll
  for (int j = 0; j < 4; ++j) {
    const int c = tid + j * 256;         // 1024 A chunks
    const int r = c >> 2;
    const int off = (c & 3) * 8;
    aSrc[j]  = A + (size_t)(row0 + r) * Kdim + off;
    aDst0[j] = (unsigned)(uintptr_t)&As[0][r][off];
  }
  const unsigned short* bSrc[2];
  unsigned bDst0[2];
#pragma unroll
  for (int j = 0; j < 2; ++j) {
    const int c = tid + j * 256;         // 512 B chunks
    const int r = c >> 2;
    const int off = (c & 3) * 8;
    int gRow;
    if constexpr (GLU) gRow = (r < 64) ? (col0 + r) : (Nout + col0 + (r - 64));
    else               gRow = col0 + r;
    bSrc[j]  = B + (size_t)gRow * Kdim + off;
    bDst0[j] = (unsigned)(uintptr_t)&Bs[0][r][off];
  }

  auto issue_tile = [&](int kt, int b) {
    const int k0 = kt * BK;
    const unsigned ofsA = (unsigned)b * AsBufBytes;
    const unsigned ofsB = (unsigned)b * BsBufBytes;
#pragma unroll
    for (int j = 0; j < 4; ++j) async_copy16(aDst0[j] + ofsA, aSrc[j] + k0);
#pragma unroll
    for (int j = 0; j < 2; ++j) async_copy16(bDst0[j] + ofsB, bSrc[j] + k0);
  };

  issue_tile(0, 0);
  wait_async0();
  __syncthreads();

  for (int kt = 0; kt < ksteps; ++kt) {
    const int cur = kt & 1;
    if (kt + 1 < ksteps) issue_tile(kt + 1, cur ^ 1);

    Frag afr[4], bfr[4];
#pragma unroll
    for (int mi = 0; mi < 4; ++mi) {
      const int m = wm0 + mi * 16 + l15;
      afr[mi].q[0] = *(const uint4*)&As[cur][m][16 * half];       // K 16h..16h+7
      afr[mi].q[1] = *(const uint4*)&As[cur][m][16 * half + 8];   // K 16h+8..
    }
#pragma unroll
    for (int ni = 0; ni < 4; ++ni) {
      int nrow;
      if constexpr (GLU) nrow = (ni < 2) ? (wn0 + ni * 16 + l15)
                                         : (64 + wn0 + (ni - 2) * 16 + l15);
      else               nrow = wn0 + ni * 16 + l15;
      bfr[ni].q[0] = *(const uint4*)&Bs[cur][nrow][16 * half];     // K 16h..
      bfr[ni].q[1] = *(const uint4*)&Bs[cur][nrow][16 * half + 8];
    }
#pragma unroll
    for (int mi = 0; mi < 4; ++mi)
#pragma unroll
      for (int ni = 0; ni < 4; ++ni)
        acc[mi][ni] = __builtin_amdgcn_wmma_f32_16x16x32_bf16(
            false, afr[mi].v, false, bfr[ni].v, (short)0, acc[mi][ni],
            false, false);

    wait_async0();
    __syncthreads();
  }

  if constexpr (GLU) {
    unsigned short* Hd = (unsigned short*)OutAll + (size_t)e * strideO;
#pragma unroll
    for (int mi = 0; mi < 4; ++mi)
#pragma unroll
      for (int ni = 0; ni < 2; ++ni) {
        const v8f a = acc[mi][ni];
        const v8f g = acc[mi][ni + 2];
        const int colg = col0 + wn0 + ni * 16 + l15;
#pragma unroll
        for (int r = 0; r < 8; ++r) {
          const int rowg = row0 + wm0 + mi * 16 + r + 8 * half;
          const float av = a[r];
          const float h = 0.5f * av * (1.f + erff(av * 0.70710678118654752f)) * g[r];
          Hd[(size_t)rowg * Nout + colg] = f32_to_bf16(h);
        }
      }
  } else {
    float* Od = (float*)OutAll + (size_t)e * strideO;
#pragma unroll
    for (int mi = 0; mi < 4; ++mi)
#pragma unroll
      for (int ni = 0; ni < 4; ++ni) {
        const v8f a = acc[mi][ni];
        const int colg = col0 + wn0 + ni * 16 + l15;
#pragma unroll
        for (int r = 0; r < 8; ++r) {
          const int rowg = row0 + wm0 + mi * 16 + r + 8 * half;
          Od[(size_t)rowg * Nout + colg] = a[r];
        }
      }
  }
}

// ---------------- 6) combine: y[s] = sum_k gate_k * outE[e_k][slot_k] ------
__global__ __launch_bounds__(256) void combine_kernel(
    const float* __restrict__ outE, const int* __restrict__ eArr,
    const int* __restrict__ slotArr, const float* __restrict__ vArr,
    float* __restrict__ y) {
  const int s = blockIdx.x;
  const int e0 = eArr[s], e1 = eArr[Ss + s];
  const int l0 = slotArr[s], l1 = slotArr[Ss + s];
  float g0 = (l0 < CAP) ? vArr[s] : 0.f;
  float g1 = (l1 < CAP) ? vArr[Ss + s] : 0.f;
  const float inv = 1.f / fmaxf(g0 + g1, 1e-9f);
  g0 *= inv; g1 *= inv;
  const float* r0 = outE + ((size_t)e0 * CAP + ((l0 < CAP) ? l0 : 0)) * Hh;
  const float* r1 = outE + ((size_t)e1 * CAP + ((l1 < CAP) ? l1 : 0)) * Hh;
  float* yr = y + (size_t)s * Hh;
  const bool k0 = (l0 < CAP), k1 = (l1 < CAP);
  for (int i = threadIdx.x; i < Hh; i += 256) {
    float acc = 0.f;
    if (k0) acc += g0 * r0[i];
    if (k1) acc += g1 * r1[i];
    yr[i] = acc;
  }
}

// ---------------- launcher ----------------
extern "C" void kernel_launch(void* const* d_in, const int* in_sizes, int n_in,
                              void* d_out, int out_size, void* d_ws, size_t ws_size,
                              hipStream_t stream) {
  (void)in_sizes; (void)n_in; (void)out_size; (void)ws_size;
  const float* x     = (const float*)d_in[0];   // [B,T,H]
  const float* Wg    = (const float*)d_in[1];   // [H,E]
  const float* W_in  = (const float*)d_in[2];   // [E,H,2F]
  const float* W_out = (const float*)d_in[3];   // [E,F,H]
  float* y = (float*)d_out;

  size_t off = 0;
  auto carve = [&](size_t bytes) -> void* {
    void* p = (char*)d_ws + off;
    off += (bytes + 255) & ~(size_t)255;
    return p;
  };
  int*   eArr    = (int*)  carve(sizeof(int)   * 2 * Ss);
  int*   slotArr = (int*)  carve(sizeof(int)   * 2 * Ss);
  float* vArr    = (float*)carve(sizeof(float) * 2 * Ss);
  unsigned short* inpB  = (unsigned short*)carve((size_t)2 * Ee * CAP * Hh);
  unsigned short* WinT  = (unsigned short*)carve((size_t)2 * Ee * TWO_F * Hh);
  unsigned short* WoutT = (unsigned short*)carve((size_t)2 * Ee * Hh * Ff);
  unsigned short* hidB  = (unsigned short*)carve((size_t)2 * Ee * CAP * Ff);
  float*          outE  = (float*)         carve(sizeof(float) * (size_t)Ee * CAP * Hh);

  zero_kernel<<<2048, 256, 0, stream>>>((uint4*)inpB,
                                        ((size_t)2 * Ee * CAP * Hh) / 16);
  router_kernel<<<Ss / 8, 256, 0, stream>>>(x, Wg, eArr, vArr);
  scan_kernel<<<1, 1024, 0, stream>>>(eArr, slotArr);
  gather_kernel<<<2 * Ss, 256, 0, stream>>>(x, eArr, slotArr, inpB);
  transpose_cast_kernel<<<dim3(TWO_F / 32, Hh / 32, Ee), dim3(32, 8), 0, stream>>>(
      W_in, WinT, Hh, TWO_F);
  transpose_cast_kernel<<<dim3(Hh / 32, Ff / 32, Ee), dim3(32, 8), 0, stream>>>(
      W_out, WoutT, Ff, Hh);
  // GEMM1 + GLU: hid[e] (CAP x 2048 bf16)
  wmma_gemm_kernel<true><<<dim3(CAP / BM, Ff / 64, Ee), 256, 0, stream>>>(
      inpB, WinT, (void*)hidB, Hh, Ff,
      (size_t)CAP * Hh, (size_t)TWO_F * Hh, (size_t)CAP * Ff);
  // GEMM2: outE[e] (CAP x 1024 f32)
  wmma_gemm_kernel<false><<<dim3(CAP / BM, Hh / 128, Ee), 256, 0, stream>>>(
      hidB, WoutT, (void*)outE, Ff, Hh,
      (size_t)CAP * Ff, (size_t)Hh * Ff, (size_t)CAP * Hh);
  combine_kernel<<<Ss, 256, 0, stream>>>(outE, eArr, slotArr, vArr, y);
}
